// GCN_21620865368322
// MI455X (gfx1250) — matrix operand order repaired and checked by reading, not compiled, not run on hardware
//
#include <hip/hip_runtime.h>
#include <stdint.h>

typedef __attribute__((ext_vector_type(16))) _Float16 v16h;
typedef __attribute__((ext_vector_type(8)))  float    v8f;

#define GN 100000
#define GE 3200000
#define GH 128
#define EPSV 1e-5f

// ---------------------------------------------------------------------------
__global__ void zero_f4(float* __restrict__ p, long n4) {
  long idx = (long)blockIdx.x * blockDim.x + threadIdx.x;
  if (idx >= n4) return;
  ((float4*)p)[idx] = make_float4(0.f, 0.f, 0.f, 0.f);
}

// x (fp32 [N,512]) -> f16 shadow concat cols 0..511 (ld 1024)
__global__ void xcast_f16(const float* __restrict__ x, _Float16* __restrict__ xh) {
  long idx = (long)blockIdx.x * blockDim.x + threadIdx.x;
  if (idx >= (long)GN * 128) return;
  long n = idx >> 7;
  int  c = (int)(idx & 127) * 4;
  float4 v = *((const float4*)(x + n * 512) + (c >> 2));
  _Float16* o = xh + n * 1024 + c;
  o[0] = (_Float16)v.x; o[1] = (_Float16)v.y;
  o[2] = (_Float16)v.z; o[3] = (_Float16)v.w;
}

// ---------------------------------------------------------------------------
// WMMA GEMM: C[M, ncols] = A[M,K] (f16, ld=lda) @ W[K,ncols] (fp32, ld=ldw)
// f16 inputs, f32 accumulation via v_wmma_f32_16x16x32_f16.
// Block = 256 threads = 8 waves; block tile 128x64; each wave owns a 32x32
// sub-tile (2x2 accumulators -> 4 WMMAs per 32-wide K step).
// A tile is staged global->LDS with GLOBAL_LOAD_ASYNC_TO_LDS_B128 (ASYNCcnt),
// no VGPR round trip. Requires K % 32 == 0; ncols guarded (tile width 64).
// Optional fused epilogue: +bias, batchnorm (eval), ReLU, f16 mirror store.
// ---------------------------------------------------------------------------
__global__ __launch_bounds__(256)
void wmma_gemm_f16(const _Float16* __restrict__ A, long lda,
                   const float* __restrict__ W, long ldw,
                   float* __restrict__ C, long ldc,
                   _Float16* __restrict__ Ch, long ldch,
                   long M, int K, int ncols,
                   const float* __restrict__ bias,
                   const float* __restrict__ bng, const float* __restrict__ bnb,
                   const float* __restrict__ bnm, const float* __restrict__ bnv,
                   int relu) {
  __shared__ _Float16 sA[128][32];  // [row][k]   16B-aligned 8-half groups
  __shared__ _Float16 sBT[64][33];  // [col][k]   transposed W tile (+1 pad)

  const int tid  = threadIdx.x;
  const int lane = tid & 31;
  const int wave = tid >> 5;
  const int wm   = wave & 3;   // row group: rows wm*32 .. wm*32+31
  const int wn   = wave >> 2;  // col group: cols wn*32 .. wn*32+31
  const int mm   = lane & 15;
  const int hh   = lane >> 4;  // half-wave select

  const long r0 = (long)blockIdx.x * 128;
  const int  c0 = blockIdx.y * 64;

  // staging assignments
  const int arow = tid >> 1;         // 0..127
  const int acol = (tid & 1) * 16;   // 0 or 16 (halfs)
  const int wrow = tid >> 3;         // 0..31 (K within tile)
  const int wcol = (tid & 7) * 8;    // 0..56
  long gar = r0 + arow;
  if (gar >= M) gar = M - 1;         // clamp: stay in-bounds, rows never stored
  const unsigned ldsA = (unsigned)(uintptr_t)&sA[arow][acol];
  const _Float16* gA  = A + gar * lda + acol;
  const bool wok = (c0 + wcol) < ncols;  // 8-col group all-in/all-out (ncols%32==0)

  v8f acc00 = {}, acc01 = {}, acc10 = {}, acc11 = {};

  for (int k0 = 0; k0 < K; k0 += 32) {
    // ---- stage A tile (128x32 f16): async DMA global -> LDS, 2x b128 ----
    {
      unsigned long long ga64 = (unsigned long long)(uintptr_t)(gA + k0);
      asm volatile(
          "global_load_async_to_lds_b128 %0, %1, off\n\t"
          "global_load_async_to_lds_b128 %0, %1, off offset:16"
          :: "v"(ldsA), "v"(ga64) : "memory");
    }
    // ---- stage W tile (32x64 fp32) transposed into sBT[col][k] ----
    {
      float4 w0 = make_float4(0.f, 0.f, 0.f, 0.f);
      float4 w1 = w0;
      if (wok) {
        const float4* wp = (const float4*)(W + (long)(k0 + wrow) * ldw + c0 + wcol);
        w0 = wp[0]; w1 = wp[1];
      }
      sBT[wcol + 0][wrow] = (_Float16)w0.x;
      sBT[wcol + 1][wrow] = (_Float16)w0.y;
      sBT[wcol + 2][wrow] = (_Float16)w0.z;
      sBT[wcol + 3][wrow] = (_Float16)w0.w;
      sBT[wcol + 4][wrow] = (_Float16)w1.x;
      sBT[wcol + 5][wrow] = (_Float16)w1.y;
      sBT[wcol + 6][wrow] = (_Float16)w1.z;
      sBT[wcol + 7][wrow] = (_Float16)w1.w;
    }
    asm volatile("s_wait_asynccnt 0x0" ::: "memory");
    __syncthreads();

    // ---- fragments per ISA 7.12.2 (wave32) ----
    // A 16x32: lane m = M; elems 0..7 -> K = hh*8 + j ; 8..15 -> K = 16 + hh*8 + j
    v16h a0, a1, b0, b1;
#pragma unroll
    for (int j = 0; j < 8; ++j) {
      a0[j]     = sA[wm * 32 + mm][hh * 8 + j];
      a0[8 + j] = sA[wm * 32 + mm][16 + hh * 8 + j];
      a1[j]     = sA[wm * 32 + 16 + mm][hh * 8 + j];
      a1[8 + j] = sA[wm * 32 + 16 + mm][16 + hh * 8 + j];
    }
    // B 32x16: lane m = N; elems 0..7 -> K = hh*8 + j ; 8..15 -> K = 16 + hh*8 + j
#pragma unroll
    for (int j = 0; j < 8; ++j) {
      b0[j]     = sBT[wn * 32 + mm][hh * 8 + j];
      b0[8 + j] = sBT[wn * 32 + mm][16 + hh * 8 + j];
      b1[j]     = sBT[wn * 32 + 16 + mm][hh * 8 + j];
      b1[8 + j] = sBT[wn * 32 + 16 + mm][16 + hh * 8 + j];
    }

    // 4 independent accumulators -> back-to-back WMMAs with disjoint D regs
    acc00 = __builtin_amdgcn_wmma_f32_16x16x32_f16(false, a0, false, b0,
                                                   (short)0, acc00, false, false);
    acc01 = __builtin_amdgcn_wmma_f32_16x16x32_f16(false, a0, false, b1,
                                                   (short)0, acc01, false, false);
    acc10 = __builtin_amdgcn_wmma_f32_16x16x32_f16(false, a1, false, b0,
                                                   (short)0, acc10, false, false);
    acc11 = __builtin_amdgcn_wmma_f32_16x16x32_f16(false, a1, false, b1,
                                                   (short)0, acc11, false, false);
    __syncthreads();
  }

  // ---- fused epilogue + store. D layout: VGPR r -> M = r + 8*hh, N = mm ----
#pragma unroll
  for (int i = 0; i < 2; ++i) {
#pragma unroll
    for (int j = 0; j < 2; ++j) {
      const v8f* accp = (i == 0) ? (j == 0 ? &acc00 : &acc01)
                                 : (j == 0 ? &acc10 : &acc11);
      const int col = c0 + wn * 32 + j * 16 + mm;
      if (col >= ncols) continue;
      float bv = bias ? bias[col] : 0.f;
      float cs = 1.f, ct = 0.f;
      if (bng) {
        float s = bng[col] * rsqrtf(bnv[col] + EPSV);
        cs = s;
        ct = bnb[col] - bnm[col] * s;
      }
#pragma unroll
      for (int r = 0; r < 8; ++r) {
        long gr = r0 + wm * 32 + i * 16 + hh * 8 + r;
        if (gr < M) {
          float v = ((*accp)[r] + bv) * cs + ct;
          if (relu) v = fmaxf(v, 0.f);
          C[gr * ldc + col] = v;
          if (Ch) Ch[gr * ldch + col] = (_Float16)v;
        }
      }
    }
  }
}

// ---------------------------------------------------------------------------
// SpMM scatter: agg[dst] += vals * hw[src]. One lane quad-of-features per
// thread (32 threads / edge). hw/agg are 51 MB each -> L2-resident atomics.
// ---------------------------------------------------------------------------
__global__ void spmm_scatter(const int* __restrict__ src,
                             const int* __restrict__ dst,
                             const float* __restrict__ vals,
                             const float* __restrict__ hw,
                             float* __restrict__ agg) {
  long idx = (long)blockIdx.x * blockDim.x + threadIdx.x;
  if (idx >= (long)GE * 32) return;
  int e = (int)(idx >> 5);
  int q = (int)(idx & 31);
  float v = vals[e];
  long s = src[e];
  long d = dst[e];
  float4 hv = *((const float4*)(hw + s * GH) + q);
  float* ap = agg + d * GH + q * 4;
  atomicAdd(ap + 0, v * hv.x);
  atomicAdd(ap + 1, v * hv.y);
  atomicAdd(ap + 2, v * hv.z);
  atomicAdd(ap + 3, v * hv.w);
}

// ---------------------------------------------------------------------------
// Segment-sum epilogue. outf = fp32 x_i slice of d_out feature region
// (ld 640, pre-offset by i*128); outh = f16 shadow concat slice (ld 1024).
// ---------------------------------------------------------------------------
__global__ void seg_bias_relu(const float* __restrict__ agg,
                              const float* __restrict__ b,
                              float* __restrict__ outf,
                              _Float16* __restrict__ outh, int relu) {
  long idx = (long)blockIdx.x * blockDim.x + threadIdx.x;
  if (idx >= (long)GN * GH) return;
  long n = idx >> 7;
  int  c = (int)(idx & 127);
  float v = agg[idx] + b[c];
  if (relu) v = fmaxf(v, 0.f);
  outf[n * 640 + c] = v;
  outh[n * 1024 + c] = (_Float16)v;
}

// ---------------------------------------------------------------------------
// Row softmax over 640 cols of S, multiply with x15, write f16 result.
// One wave32 per node (20 cols/lane), shfl_xor reductions.
// ---------------------------------------------------------------------------
__global__ __launch_bounds__(256)
void attn_softmax_mul(const float* __restrict__ S, const float* __restrict__ x15,
                      _Float16* __restrict__ ah) {
  int wave = threadIdx.x >> 5;
  int lane = threadIdx.x & 31;
  long n = (long)blockIdx.x * 8 + wave;
  if (n >= GN) return;
  const float* row = S + n * 640;
  float v[20];
  float m = -3.4e38f;
#pragma unroll
  for (int t = 0; t < 20; ++t) {
    v[t] = row[lane + 32 * t];
    m = fmaxf(m, v[t]);
  }
  for (int off = 16; off; off >>= 1) m = fmaxf(m, __shfl_xor(m, off, 32));
  float sum = 0.f;
#pragma unroll
  for (int t = 0; t < 20; ++t) {
    v[t] = __expf(v[t] - m);
    sum += v[t];
  }
  for (int off = 16; off; off >>= 1) sum += __shfl_xor(sum, off, 32);
  float inv = 1.f / sum;
  const float* xr = x15 + n * 640;
  _Float16* orow = ah + n * 640;
#pragma unroll
  for (int t = 0; t < 20; ++t) {
    int c = lane + 32 * t;
    orow[c] = (_Float16)(xr[c] * (v[t] * inv));
  }
}

// ---------------------------------------------------------------------------
// Head: logits = h2[N,32] @ H3w[32,10] + H3b, then log_softmax. 1 thread/node.
// ---------------------------------------------------------------------------
__global__ void head_logsoftmax(const float* __restrict__ h2,
                                const float* __restrict__ W3,
                                const float* __restrict__ b3,
                                float* __restrict__ out) {
  long n = (long)blockIdx.x * blockDim.x + threadIdx.x;
  if (n >= GN) return;
  float lg[10];
#pragma unroll
  for (int c = 0; c < 10; ++c) lg[c] = b3[c];
#pragma unroll
  for (int i = 0; i < 32; ++i) {
    float hv = h2[n * 32 + i];
#pragma unroll
    for (int c = 0; c < 10; ++c) lg[c] += hv * W3[i * 10 + c];
  }
  float m = lg[0];
#pragma unroll
  for (int c = 1; c < 10; ++c) m = fmaxf(m, lg[c]);
  float s = 0.f;
#pragma unroll
  for (int c = 0; c < 10; ++c) s += __expf(lg[c] - m);
  float ls = __logf(s);
#pragma unroll
  for (int c = 0; c < 10; ++c) out[n * 10 + c] = lg[c] - m - ls;
}

// ---------------------------------------------------------------------------
extern "C" void kernel_launch(void* const* d_in, const int* in_sizes, int n_in,
                              void* d_out, int out_size, void* d_ws, size_t ws_size,
                              hipStream_t stream) {
  (void)in_sizes; (void)n_in; (void)out_size; (void)ws_size;
  const float* x    = (const float*)d_in[0];
  const int*   esrc = (const int*)d_in[1];
  const int*   edst = (const int*)d_in[2];
  const float* ev   = (const float*)d_in[3];
  const float* Wl[5], *bl[5];
  for (int i = 0; i < 5; ++i) {
    Wl[i] = (const float*)d_in[4 + 2 * i];
    bl[i] = (const float*)d_in[5 + 2 * i];
  }
  const float* Wa  = (const float*)d_in[14];
  const float* ba  = (const float*)d_in[15];
  const float* H1w = (const float*)d_in[16];
  const float* H1b = (const float*)d_in[17];
  const float* bn1g = (const float*)d_in[18];
  const float* bn1b = (const float*)d_in[19];
  const float* bn1m = (const float*)d_in[20];
  const float* bn1v = (const float*)d_in[21];
  const float* H2w = (const float*)d_in[22];
  const float* H2b = (const float*)d_in[23];
  const float* bn2g = (const float*)d_in[24];
  const float* bn2b = (const float*)d_in[25];
  const float* bn2m = (const float*)d_in[26];
  const float* bn2v = (const float*)d_in[27];
  const float* H3w = (const float*)d_in[28];
  const float* H3b = (const float*)d_in[29];

  // workspace carve-up (all offsets multiples of 16 bytes)
  char* base = (char*)d_ws;
  _Float16* XCATH = (_Float16*)base;                         // [N,1024] f16 concat
  float*    HW    = (float*)(base + (size_t)GN * 1024 * 2);  // [N,128] f32
  float*    AGG   = HW  + (size_t)GN * 128;                  // [N,128] f32
  float*    ATT   = AGG + (size_t)GN * 128;                  // [N,640] f32 (S)
  _Float16* ATTH  = (_Float16*)(ATT + (size_t)GN * 640);     // [N,640] f16 (a)
  _Float16* HWH   = ATTH + (size_t)GN * 640;                 // [N,128] f16 (h1)
  float*    H2B   = (float*)(HWH + (size_t)GN * 128);        // [N,32]  f32
  float*    OUT   = (float*)d_out;            // N*10 log-probs, then N*640 feats
  float*    OUTF  = OUT + (size_t)GN * 10;    // feature region = x15 fp32, ld 640

  dim3 blk(256);
  long gemmGX = (GN + 127) / 128;

  // x -> f16 concat cols 0..511
  {
    long tot = (long)GN * 128;
    xcast_f16<<<dim3((tot + 255) / 256), blk, 0, stream>>>(x, XCATH);
  }

  // 5 dense-GCN layers: hw = concat[:, :din] @ W ; agg = A_sp @ hw ; +b (+relu)
  for (int i = 0; i < 5; ++i) {
    int K = 512 + i * 128;
    wmma_gemm_f16<<<dim3(gemmGX, 2), blk, 0, stream>>>(
        XCATH, 1024, Wl[i], 128, HW, 128, nullptr, 0, GN, K, 128,
        nullptr, nullptr, nullptr, nullptr, nullptr, 0);
    {
      long n4 = (long)GN * 128 / 4;
      zero_f4<<<dim3((n4 + 255) / 256), blk, 0, stream>>>(AGG, n4);
    }
    {
      long tot = (long)GE * 32;
      spmm_scatter<<<dim3((tot + 255) / 256), blk, 0, stream>>>(esrc, edst, ev, HW, AGG);
    }
    {
      long tot = (long)GN * 128;
      seg_bias_relu<<<dim3((tot + 255) / 256), blk, 0, stream>>>(
          AGG, bl[i], OUTF + i * 128, XCATH + 512 + i * 128, (i == 0) ? 1 : 0);
    }
  }

  // attention: S = x15 @ Wa + ba ; a = x15 * softmax(S) -> ATTH (f16)
  wmma_gemm_f16<<<dim3(gemmGX, 10), blk, 0, stream>>>(
      XCATH + 512, 1024, Wa, 640, ATT, 640, nullptr, 0, GN, 640, 640,
      ba, nullptr, nullptr, nullptr, nullptr, 0);
  attn_softmax_mul<<<dim3((GN + 7) / 8), blk, 0, stream>>>(ATT, OUTF, ATTH);

  // H1: relu(bn1(a @ H1w + H1b)) -> HW (fp32) + HWH (f16 mirror for H2)
  wmma_gemm_f16<<<dim3(gemmGX, 2), blk, 0, stream>>>(
      ATTH, 640, H1w, 128, HW, 128, HWH, 128, GN, 640, 128,
      H1b, bn1g, bn1b, bn1m, bn1v, 1);

  // H2: relu(bn2(h1 @ H2w + H2b)) -> H2B (ncols=32, col-guarded 64-wide tile)
  wmma_gemm_f16<<<dim3(gemmGX, 1), blk, 0, stream>>>(
      HWH, 128, H2w, 32, H2B, 32, nullptr, 0, GN, 128, 32,
      H2b, bn2g, bn2b, bn2m, bn2v, 1);

  // head + log_softmax -> OUT[0 : N*10)
  head_logsoftmax<<<dim3((GN + 255) / 256), blk, 0, stream>>>(H2B, H3w, H3b, OUT);
}
